// GIN_67877663146445
// MI455X (gfx1250) — compile-verified, matrix operand order checked by reading, not run on hardware
//
#include <hip/hip_runtime.h>
#include <hip/hip_bf16.h>
#include <math.h>

#define NNODES   100000
#define NPAD     100096            // ceil(NNODES/128)*128 -> branch-free WMMA tiles
#define HID      128
#define NGRAPHS  512
#define INCH     7
#define BN_EPS   1e-5f

typedef __attribute__((ext_vector_type(16))) __bf16 v16bf;
typedef __attribute__((ext_vector_type(8)))  float  v8f;

// ---------------------------------------------------------------- utilities

__global__ __launch_bounds__(256) void gin_fill(float* __restrict__ p, int n, float v) {
    int i = blockIdx.x * 256 + threadIdx.x;
    if (i < n) p[i] = v;
}

__device__ __forceinline__ void atomicMaxFloat(float* addr, float value) {
    // monotone float<->int ordering trick; requires init to -inf (0xFF800000)
    if (value >= 0.0f)
        atomicMax((int*)addr, __float_as_int(value));
    else
        atomicMin((unsigned int*)addr, __float_as_uint(value));
}

// ------------------------------------------------------- edge aggregation

// 7-channel layer-0 aggregation: one thread per edge
__global__ __launch_bounds__(256) void gin_edge_agg7(
    const float* __restrict__ h, const int* __restrict__ src,
    const int* __restrict__ dst, float* __restrict__ agg, int nedges)
{
    int e = blockIdx.x * 256 + threadIdx.x;
    if (e >= nedges) return;
    const int s = src[e], d = dst[e];
    const float* hs = h + (size_t)s * INCH;
    float* ad = agg + (size_t)d * INCH;
#pragma unroll
    for (int c = 0; c < INCH; ++c) atomicAdd(&ad[c], hs[c]);
}

// 128-channel aggregation: 32 threads per edge, float4 per thread
__global__ __launch_bounds__(256) void gin_edge_agg128(
    const float* __restrict__ h, const int* __restrict__ src,
    const int* __restrict__ dst, float* __restrict__ agg, int nedges)
{
    long t = (long)blockIdx.x * 256 + threadIdx.x;
    long e = t >> 5;
    if (e >= nedges) return;
    const int fg = (int)(t & 31) * 4;
    const int s = src[e], d = dst[e];
    const float4 v = *(const float4*)(h + (size_t)s * HID + fg);
    float* ad = agg + (size_t)d * HID + fg;
    atomicAdd(&ad[0], v.x);
    atomicAdd(&ad[1], v.y);
    atomicAdd(&ad[2], v.z);
    atomicAdd(&ad[3], v.w);
}

// ------------------------------------------------------------- GEMMs

// layer-0 first linear: (N x 7) @ (7 x 128) + b   (tiny K -> VALU)
__global__ __launch_bounds__(256) void gin_gemm7(
    const float* __restrict__ Z, const float* __restrict__ W,
    const float* __restrict__ b, float* __restrict__ T, int nrows)
{
    int idx = blockIdx.x * 256 + threadIdx.x;
    if (idx >= nrows * HID) return;
    const int n = idx >> 7, c = idx & 127;
    const float* zr = Z + (size_t)n * INCH;
    float acc = b[c];
#pragma unroll
    for (int k = 0; k < INCH; ++k) acc += zr[k] * W[k * HID + c];
    T[idx] = acc;
}

__device__ __forceinline__ v16bf pack16(float4 a, float4 b, float4 c, float4 d) {
    v16bf v;
    v[0]  = (__bf16)a.x; v[1]  = (__bf16)a.y; v[2]  = (__bf16)a.z; v[3]  = (__bf16)a.w;
    v[4]  = (__bf16)b.x; v[5]  = (__bf16)b.y; v[6]  = (__bf16)b.z; v[7]  = (__bf16)b.w;
    v[8]  = (__bf16)c.x; v[9]  = (__bf16)c.y; v[10] = (__bf16)c.z; v[11] = (__bf16)c.w;
    v[12] = (__bf16)d.x; v[13] = (__bf16)d.y; v[14] = (__bf16)d.z; v[15] = (__bf16)d.w;
    return v;
}

// (NPAD x 128) @ (128 x 128) + b with bf16 WMMA, f32 accumulate.
// Block = 256 threads = 8 waves; wave w owns cols [16w,16w+16); B is register-
// resident across 8 row-tiles (128 rows) per block. A rows are padded so all
// loads / WMMAs / stores are unconditional (no exec-mask churn).
__global__ __launch_bounds__(256) void gin_gemm128_wmma(
    const float* __restrict__ A, const float* __restrict__ W,
    const float* __restrict__ bias, float* __restrict__ T)
{
    const int wave = threadIdx.x >> 5;
    const int lane = threadIdx.x & 31;
    const int half = lane >> 4;
    const int lm   = lane & 15;
    const int col  = wave * 16 + lm;

    // B tile (16-bit B layout): lane holds column `col`; element e of chunk c
    // covers K = c*32 + ((e<8)?0:16) + half*8 + (e&7)
    v16bf B[4];
#pragma unroll
    for (int c = 0; c < 4; ++c) {
#pragma unroll
        for (int e = 0; e < 16; ++e) {
            const int kk = c * 32 + ((e < 8) ? 0 : 16) + half * 8 + (e & 7);
            B[c][e] = (__bf16)W[kk * HID + col];
        }
    }
    const float bv = bias[col];

    const size_t rowbase = (size_t)blockIdx.x * 128;
#pragma unroll 1
    for (int mt = 0; mt < 8; ++mt) {
        const size_t row0 = rowbase + (size_t)mt * 16;
        // A tile (16-bit A layout): lane holds row row0+lm; per 32-K chunk the
        // lane's 16 elements are two contiguous 8-float runs -> 4 x b128 loads
        const float4* a4 = (const float4*)(A + (row0 + lm) * HID);
        v16bf Ar[4];
#pragma unroll
        for (int c = 0; c < 4; ++c) {
            const float4 p0 = a4[c * 8 + half * 2 + 0];
            const float4 p1 = a4[c * 8 + half * 2 + 1];
            const float4 p2 = a4[c * 8 + 4 + half * 2 + 0];
            const float4 p3 = a4[c * 8 + 4 + half * 2 + 1];
            Ar[c] = pack16(p0, p1, p2, p3);
        }

        v8f acc = {};
        acc = __builtin_amdgcn_wmma_f32_16x16x32_bf16(false, Ar[0], false, B[0], (short)0, acc, false, false);
        acc = __builtin_amdgcn_wmma_f32_16x16x32_bf16(false, Ar[1], false, B[1], (short)0, acc, false, false);
        acc = __builtin_amdgcn_wmma_f32_16x16x32_bf16(false, Ar[2], false, B[2], (short)0, acc, false, false);
        acc = __builtin_amdgcn_wmma_f32_16x16x32_bf16(false, Ar[3], false, B[3], (short)0, acc, false, false);

        // C/D layout: VGPR r holds row (r + 8*half), col = lane%16
        float* trow = T + row0 * HID + col;
#pragma unroll
        for (int r = 0; r < 8; ++r)
            trow[(size_t)(r + half * 8) * HID] = acc[r] + bv;
    }
}

// -------------------------------------------------------- BatchNorm (2-pass)

__global__ __launch_bounds__(256) void gin_colstats(
    const float* __restrict__ T, float* __restrict__ stats, int nrows)
{
    const int c = threadIdx.x & 127;
    const int sub = threadIdx.x >> 7;      // 2 threads per column
    long r = (long)blockIdx.x * 512 + sub;
    long rend = (long)(blockIdx.x + 1) * 512;
    if (rend > nrows) rend = nrows;
    float s = 0.0f, s2 = 0.0f;
    for (; r < rend; r += 2) {
        const float v = T[r * HID + c];
        s += v; s2 += v * v;
    }
    __shared__ float sh[256];
    sh[threadIdx.x] = s;  __syncthreads();
    if (sub == 0) atomicAdd(&stats[c], sh[c] + sh[c + 128]);
    __syncthreads();
    sh[threadIdx.x] = s2; __syncthreads();
    if (sub == 0) atomicAdd(&stats[c + 128], sh[c] + sh[c + 128]);
}

__global__ __launch_bounds__(128) void gin_bnfin(
    const float* __restrict__ stats, const float* __restrict__ g,
    const float* __restrict__ be, float* __restrict__ coef, float invn)
{
    const int c = threadIdx.x;
    const float m  = stats[c] * invn;
    const float v  = stats[c + 128] * invn - m * m;
    const float sc = g[c] * rsqrtf(v + BN_EPS);
    coef[c]       = sc;
    coef[c + 128] = be[c] - m * sc;
}

__global__ __launch_bounds__(256) void gin_bnrelu(
    const float* __restrict__ T, const float* __restrict__ coef,
    float* __restrict__ out, int nrows)
{
    int idx = blockIdx.x * 256 + threadIdx.x;
    if (idx >= nrows * HID) return;
    const int c = idx & 127;
    const float y = T[idx] * coef[c] + coef[c + 128];
    out[idx] = fmaxf(y, 0.0f);
}

// ------------------------------------------------------------- pooling/head

template <int C>
__global__ __launch_bounds__(256) void gin_pool(
    const float* __restrict__ h, const int* __restrict__ batch,
    float* __restrict__ pool, int nrows)
{
    long idx = (long)blockIdx.x * 256 + threadIdx.x;
    if (idx >= (long)nrows * C) return;
    const long n = idx / C;                 // C is compile-time -> mul/shift
    const int  c = (int)(idx - n * C);
    atomicMaxFloat(&pool[(long)batch[n] * HID + c], h[idx]);
}

struct HeadAll {
    const float* W0[4]; const float* b0[4];
    const float* W1[4]; const float* b1[4];
};

__global__ __launch_bounds__(128) void gin_heads(
    const float* __restrict__ pool, HeadAll hp, float* __restrict__ out)
{
    const int g = blockIdx.x;
    const int c = threadIdx.x;
    __shared__ float hid[HID];
    __shared__ float sc[2];
    if (c < 2) sc[c] = 0.0f;
#pragma unroll 1
    for (int h = 0; h < 4; ++h) {
        const int din = (h == 0) ? INCH : HID;
        const float* p = pool + (size_t)h * NGRAPHS * HID + (size_t)g * HID;
        float acc = hp.b0[h][c];
        for (int k = 0; k < din; ++k) {
            float x = p[k];
            x = __builtin_isfinite(x) ? x : 0.0f;   // guard empty graphs (-inf)
            acc += x * hp.W0[h][k * HID + c];
        }
        __syncthreads();
        hid[c] = fmaxf(acc, 0.0f);
        __syncthreads();
        if (c < 2) {
            float s = hp.b1[h][c];
            for (int k = 0; k < HID; ++k) s += hid[k] * hp.W1[h][k * 2 + c];
            sc[c] += s;
        }
    }
    __syncthreads();
    if (c < 2) out[g * 2 + c] = sc[c];
}

// ---------------------------------------------------------------- driver

extern "C" void kernel_launch(void* const* d_in, const int* in_sizes, int n_in,
                              void* d_out, int out_size, void* d_ws, size_t ws_size,
                              hipStream_t stream)
{
    (void)n_in; (void)out_size; (void)ws_size;

    const float* x    = (const float*)d_in[0];
    const int*   ei   = (const int*)d_in[1];
    const int nedges  = in_sizes[1] / 2;
    const int* src    = ei;
    const int* dst    = ei + nedges;
    const int* batch  = (const int*)d_in[2];

    // conv_params: per layer {W0,b0,g0,be0,W1,b1} starting at index 3
    const float *cW0[3], *cb0[3], *cg0[3], *cbe0[3], *cW1[3], *cb1[3];
    for (int i = 0; i < 3; ++i) {
        cW0[i]  = (const float*)d_in[3 + 6*i + 0];
        cb0[i]  = (const float*)d_in[3 + 6*i + 1];
        cg0[i]  = (const float*)d_in[3 + 6*i + 2];
        cbe0[i] = (const float*)d_in[3 + 6*i + 3];
        cW1[i]  = (const float*)d_in[3 + 6*i + 4];
        cb1[i]  = (const float*)d_in[3 + 6*i + 5];
    }
    // bn_params: (g,b) pairs at index 21
    const float *bng[3], *bnb[3];
    for (int i = 0; i < 3; ++i) {
        bng[i] = (const float*)d_in[21 + 2*i + 0];
        bnb[i] = (const float*)d_in[21 + 2*i + 1];
    }
    // head_params: {W0,b0,W1,b1} at index 27
    HeadAll hp;
    for (int i = 0; i < 4; ++i) {
        hp.W0[i] = (const float*)d_in[27 + 4*i + 0];
        hp.b0[i] = (const float*)d_in[27 + 4*i + 1];
        hp.W1[i] = (const float*)d_in[27 + 4*i + 2];
        hp.b1[i] = (const float*)d_in[27 + 4*i + 3];
    }

    // workspace layout (floats); node buffers padded to NPAD rows so the WMMA
    // GEMM is fully branch-free (tail rows are dead data, never consumed)
    float* H0 = (float*)d_ws;                      // node features ping
    float* H1 = H0 + (size_t)NPAD * HID;           // node features pong
    float* ZB = H1 + (size_t)NPAD * HID;           // z = h + agg / activations
    float* TB = ZB + (size_t)NPAD * HID;           // GEMM output
    float* Z7 = TB + (size_t)NPAD * HID;           // layer-0 z (7 ch)
    float* PO = Z7 + (size_t)NNODES * INCH;        // 4 pooled reps, stride 128
    float* ST = PO + (size_t)4 * NGRAPHS * HID;    // [256] sum|sumsq
    float* CF = ST + 256;                          // [256] scale|shift

    const int   elemBlocks = (NNODES * HID + 255) / 256;
    const int   gemmBlocks = NPAD / 128;           // 782, all tiles full
    const int   statBlocks = (NNODES + 511) / 512;
    const float invn       = 1.0f / (float)NNODES;

    // init pooled reps to -inf
    gin_fill<<<(4 * NGRAPHS * HID + 255) / 256, 256, 0, stream>>>(
        PO, 4 * NGRAPHS * HID, -INFINITY);

    // rep 0 pooling (raw x, 7 ch)
    gin_pool<INCH><<<(NNODES * INCH + 255) / 256, 256, 0, stream>>>(x, batch, PO, NNODES);

    // ---------------- layer 0 (7 -> 128) ----------------
    hipMemcpyAsync(Z7, x, sizeof(float) * (size_t)NNODES * INCH,
                   hipMemcpyDeviceToDevice, stream);
    gin_edge_agg7<<<(nedges + 255) / 256, 256, 0, stream>>>(x, src, dst, Z7, nedges);

    gin_fill<<<1, 256, 0, stream>>>(ST, 256, 0.0f);
    gin_gemm7<<<elemBlocks, 256, 0, stream>>>(Z7, cW0[0], cb0[0], TB, NNODES);
    gin_colstats<<<statBlocks, 256, 0, stream>>>(TB, ST, NNODES);
    gin_bnfin<<<1, 128, 0, stream>>>(ST, cg0[0], cbe0[0], CF, invn);
    gin_bnrelu<<<elemBlocks, 256, 0, stream>>>(TB, CF, ZB, NNODES);

    gin_fill<<<1, 256, 0, stream>>>(ST, 256, 0.0f);
    gin_gemm128_wmma<<<gemmBlocks, 256, 0, stream>>>(ZB, cW1[0], cb1[0], TB);
    gin_colstats<<<statBlocks, 256, 0, stream>>>(TB, ST, NNODES);
    gin_bnfin<<<1, 128, 0, stream>>>(ST, bng[0], bnb[0], CF, invn);
    gin_bnrelu<<<elemBlocks, 256, 0, stream>>>(TB, CF, H0, NNODES);
    gin_pool<HID><<<elemBlocks, 256, 0, stream>>>(H0, batch, PO + (size_t)1 * NGRAPHS * HID, NNODES);

    // ---------------- layers 1,2 (128 -> 128) ----------------
    float* hc = H0;
    float* hn = H1;
    const unsigned aggBlocks = (unsigned)(((long)nedges * 32 + 255) / 256);
    for (int L = 1; L < 3; ++L) {
        hipMemcpyAsync(ZB, hc, sizeof(float) * (size_t)NNODES * HID,
                       hipMemcpyDeviceToDevice, stream);
        gin_edge_agg128<<<aggBlocks, 256, 0, stream>>>(hc, src, dst, ZB, nedges);

        gin_fill<<<1, 256, 0, stream>>>(ST, 256, 0.0f);
        gin_gemm128_wmma<<<gemmBlocks, 256, 0, stream>>>(ZB, cW0[L], cb0[L], TB);
        gin_colstats<<<statBlocks, 256, 0, stream>>>(TB, ST, NNODES);
        gin_bnfin<<<1, 128, 0, stream>>>(ST, cg0[L], cbe0[L], CF, invn);
        gin_bnrelu<<<elemBlocks, 256, 0, stream>>>(TB, CF, ZB, NNODES);

        gin_fill<<<1, 256, 0, stream>>>(ST, 256, 0.0f);
        gin_gemm128_wmma<<<gemmBlocks, 256, 0, stream>>>(ZB, cW1[L], cb1[L], TB);
        gin_colstats<<<statBlocks, 256, 0, stream>>>(TB, ST, NNODES);
        gin_bnfin<<<1, 128, 0, stream>>>(ST, bng[L], bnb[L], CF, invn);
        gin_bnrelu<<<elemBlocks, 256, 0, stream>>>(TB, CF, hn, NNODES);

        gin_pool<HID><<<elemBlocks, 256, 0, stream>>>(
            hn, batch, PO + (size_t)(L + 1) * NGRAPHS * HID, NNODES);

        float* t = hc; hc = hn; hn = t;
    }

    // ---------------- heads -> score ----------------
    gin_heads<<<NGRAPHS, 128, 0, stream>>>(PO, hp, (float*)d_out);
}
